// JointLearningMemoryModel_5755256177376
// MI455X (gfx1250) — compile-verified, matrix operand order checked by reading, not run on hardware
//
#include <hip/hip_runtime.h>
#include <hip/hip_bf16.h>
#include <math.h>

// ---- problem dims (from reference) ----
#define H     768
#define NH    12
#define HD    64
#define FF    3072
#define NSEG  8
#define SLEN  512
#define MROWS 511          // number of mentions
#define NROWS 512          // M+1 == padded row count (multiple of 16)
#define HHALF 384
#define HQ    192

// ---- WMMA types (gfx1250, wave32) ----
typedef __attribute__((ext_vector_type(16))) __bf16       v16bf;
typedef __attribute__((ext_vector_type(8)))  float        v8f;
typedef __attribute__((ext_vector_type(4)))  unsigned int u32x4;  // trivial 128-bit chunk

union FragBF { v16bf v; u32x4 u[2]; };

static __device__ __forceinline__ __bf16 f2bf(float f) {
  union { float f; unsigned u; } in; in.f = f;
  unsigned u = in.u;
  unsigned r = u + 0x7fffu + ((u >> 16) & 1u);   // round-to-nearest-even
  union { unsigned short s; __bf16 b; } o; o.s = (unsigned short)(r >> 16);
  return o.b;
}

// ---------------------------------------------------------------------------
// bf16 WMMA GEMM with 4-way N-tiling: C[M,N] = act(A[M,K] @ B[K,N] + bias)
// A row-major bf16, Bt = B transposed, row-major [N,K] bf16.
// One wave computes a 16x64 C strip: one A fragment feeds 4 WMMAs per K-step
// (4x register reuse of A), 4 independent accumulators hide WMMA latency.
// Requires N % 64 == 0 (all N here are 768 or 3072).
// act: 0 = none, 1 = exact GELU (erf), 2 = ReLU
// ---------------------------------------------------------------------------
__global__ void __launch_bounds__(32)
gemm_wmma(const __bf16* __restrict__ A, const __bf16* __restrict__ Bt,
          const float* __restrict__ bias, float* __restrict__ C,
          int M, int N, int K, int act) {
  const int lane = threadIdx.x;
  const int r16  = lane & 15;
  const int half = lane >> 4;
  const int rowA  = blockIdx.y * 16 + r16;    // A-frag: lane holds row (lane&15)
  const int cbase = blockIdx.x * 64;          // 4 consecutive 16-col B tiles
  const __bf16* a0 = A + (size_t)rowA * K;
  const __bf16* bp[4];
#pragma unroll
  for (int t = 0; t < 4; ++t)
    bp[t] = Bt + (size_t)(cbase + t * 16 + r16) * K;   // B-frag: lane holds col

  v8f acc[4];
#pragma unroll
  for (int t = 0; t < 4; ++t) acc[t] = (v8f){};

  for (int kb = 0; kb < K; kb += 32) {
    FragBF fa;
    __builtin_prefetch(a0 + kb + 256, 0, 3);
    fa.u[0] = *reinterpret_cast<const u32x4*>(a0 + kb +      half * 8);
    fa.u[1] = *reinterpret_cast<const u32x4*>(a0 + kb + 16 + half * 8);
    FragBF fb[4];
#pragma unroll
    for (int t = 0; t < 4; ++t) {
      fb[t].u[0] = *reinterpret_cast<const u32x4*>(bp[t] + kb +      half * 8);
      fb[t].u[1] = *reinterpret_cast<const u32x4*>(bp[t] + kb + 16 + half * 8);
    }
#pragma unroll
    for (int t = 0; t < 4; ++t)
      acc[t] = __builtin_amdgcn_wmma_f32_16x16x32_bf16(false, fa.v, false, fb[t].v,
                                                       (short)0, acc[t], false, false);
  }

  const int rbase = blockIdx.y * 16 + half * 8;   // D rows = (lane>>4)*8 + ii
#pragma unroll
  for (int t = 0; t < 4; ++t) {
    const int col  = cbase + t * 16 + r16;        // D col = lane&15 within tile
    const float bb = bias ? bias[col] : 0.f;
#pragma unroll
    for (int ii = 0; ii < 8; ++ii) {
      float vv = acc[t][ii] + bb;
      if (act == 1)      vv = 0.5f * vv * (1.f + erff(vv * 0.70710678118654752f));
      else if (act == 2) vv = fmaxf(vv, 0.f);
      C[(size_t)(rbase + ii) * N + col] = vv;
    }
  }
}

// ---------------------------------------------------------------------------
// Fused pairwise scorer:
//   out[i,j] = pW3 . relu(pW2^T relu(pa[j]+pb[i]) + pb2) + pb3 + ms[i] + ms[j]
//              + (j>=i ? -1e4 : 0)
// One wave per (i, 16-wide j-tile). The 16x768 h1 tile lives only in LDS
// (bf16, 24KB); pW2T streams from L2. h1 [512,512,768] is never materialized.
// 4-way N-tiling: each LDS A fragment feeds 4 WMMAs (6 outer chunks of 64).
// ---------------------------------------------------------------------------
__global__ void __launch_bounds__(32)
pair_kernel(const float* __restrict__ pa, const float* __restrict__ pb,
            const __bf16* __restrict__ pW2T, const float* __restrict__ pb2,
            const float* __restrict__ pW3, const float* __restrict__ pb3,
            const float* __restrict__ ms, float* __restrict__ out) {
  const int jt   = blockIdx.x;      // 0..31
  const int i    = blockIdx.y;      // 0..511
  const int lane = threadIdx.x;
  const int r16  = lane & 15;
  const int half = lane >> 4;

  __shared__ __bf16 Atile[16 * H];      // 24 KB
  __shared__ float  redbuf[16][17];

  // build h1 tile = relu(pa[j] + pb[i]) in bf16
  const float* pbrow = pb + (size_t)i * H;
  for (int jr = 0; jr < 16; ++jr) {
    const float* parow = pa + (size_t)(jt * 16 + jr) * H;
    for (int hh = lane; hh < H; hh += 32)
      Atile[jr * H + hh] = f2bf(fmaxf(parow[hh] + pbrow[hh], 0.f));
  }
  __syncthreads();

  const __bf16* a0 = Atile + r16 * H;
  float contrib[8];
#pragma unroll
  for (int ii = 0; ii < 8; ++ii) contrib[ii] = 0.f;

  for (int nc = 0; nc < HHALF / 64; ++nc) {     // 6 chunks of 64 pW2 columns
    const __bf16* bp[4];
#pragma unroll
    for (int t = 0; t < 4; ++t)
      bp[t] = pW2T + (size_t)(nc * 64 + t * 16 + r16) * H;

    v8f acc[4];
#pragma unroll
    for (int t = 0; t < 4; ++t) acc[t] = (v8f){};

    for (int kb = 0; kb < H; kb += 32) {
      FragBF fa;
      fa.u[0] = *reinterpret_cast<const u32x4*>(a0 + kb +      half * 8);
      fa.u[1] = *reinterpret_cast<const u32x4*>(a0 + kb + 16 + half * 8);
      FragBF fb[4];
#pragma unroll
      for (int t = 0; t < 4; ++t) {
        __builtin_prefetch(bp[t] + kb + 256, 0, 3);
        fb[t].u[0] = *reinterpret_cast<const u32x4*>(bp[t] + kb +      half * 8);
        fb[t].u[1] = *reinterpret_cast<const u32x4*>(bp[t] + kb + 16 + half * 8);
      }
#pragma unroll
      for (int t = 0; t < 4; ++t)
        acc[t] = __builtin_amdgcn_wmma_f32_16x16x32_bf16(false, fa.v, false, fb[t].v,
                                                         (short)0, acc[t], false, false);
    }
#pragma unroll
    for (int t = 0; t < 4; ++t) {
      const int col  = nc * 64 + t * 16 + r16;
      const float bb = pb2[col], w3 = pW3[col];
#pragma unroll
      for (int ii = 0; ii < 8; ++ii) {
        float h2 = fmaxf(acc[t][ii] + bb, 0.f);
        contrib[ii] += h2 * w3;                  // contract with pW3 in-register
      }
    }
  }

  // cross-lane reduction: row j's partials live in 16 lanes (one per col)
#pragma unroll
  for (int ii = 0; ii < 8; ++ii) redbuf[half * 8 + ii][r16] = contrib[ii];
  __syncthreads();
  if (lane < 16) {
    float s = 0.f;
#pragma unroll
    for (int c = 0; c < 16; ++c) s += redbuf[lane][c];
    const int jg = jt * 16 + lane;
    float val = s + pb3[0] + ms[i] + ms[jg] + ((jg >= i) ? -10000.f : 0.f);
    out[(size_t)i * NROWS + jg] = val;
  }
}

// ---------------------------------------------------------------------------
// Small VALU kernels
// ---------------------------------------------------------------------------
__global__ void gather_kernel(const float* __restrict__ seq,
                              const float* __restrict__ spk_emb,
                              const int* __restrict__ mseg, const int* __restrict__ mstart,
                              const int* __restrict__ mend, const int* __restrict__ spk_ids,
                              float* __restrict__ m) {
  const int row = blockIdx.x, tid = threadIdx.x;
  if (row >= MROWS) {                       // zero the padding row
    for (int idx = tid; idx < H; idx += 256) m[(size_t)row * H + idx] = 0.f;
    return;
  }
  const int seg = mseg[row], st = mstart[row], en = mend[row];
  const int sid = spk_ids[seg * SLEN + st];
  const float* s1 = seq + ((size_t)seg * SLEN + st) * H;
  const float* s2 = seq + ((size_t)seg * SLEN + en) * H;
  const float* sp = spk_emb + (size_t)sid * H;
  for (int idx = tid; idx < H; idx += 256)
    m[(size_t)row * H + idx] = s1[idx] + s2[idx] + sp[idx];
}

__global__ void attn_kernel(const float* __restrict__ q, const float* __restrict__ k,
                            const float* __restrict__ v, float* __restrict__ ctx) {
  const int h = blockIdx.x, i = blockIdx.y, tid = threadIdx.x;   // block = 128
  __shared__ float qh[HD];
  __shared__ float sc[NROWS];
  __shared__ float red[128];
  if (i >= MROWS) {
    for (int d = tid; d < HD; d += 128) ctx[(size_t)i * H + h * HD + d] = 0.f;
    return;
  }
  if (tid < HD) qh[tid] = q[(size_t)i * H + h * HD + tid];
  __syncthreads();
  float lmax = -1e30f;
  for (int j = tid; j < MROWS; j += 128) {
    const float* kr = k + (size_t)j * H + h * HD;
    float s = 0.f;
#pragma unroll 8
    for (int d = 0; d < HD; ++d) s += qh[d] * kr[d];
    s *= 0.125f;                 // 1/sqrt(64)
    sc[j] = s;
    lmax = fmaxf(lmax, s);
  }
  red[tid] = lmax; __syncthreads();
  for (int s = 64; s > 0; s >>= 1) { if (tid < s) red[tid] = fmaxf(red[tid], red[tid + s]); __syncthreads(); }
  const float mx = red[0];
  __syncthreads();
  float lsum = 0.f;
  for (int j = tid; j < MROWS; j += 128) { float e = __expf(sc[j] - mx); sc[j] = e; lsum += e; }
  red[tid] = lsum; __syncthreads();
  for (int s = 64; s > 0; s >>= 1) { if (tid < s) red[tid] += red[tid + s]; __syncthreads(); }
  const float inv = 1.f / red[0];
  __syncthreads();
  for (int d = tid; d < HD; d += 128) {
    float s = 0.f;
    for (int j = 0; j < MROWS; ++j) s += sc[j] * v[(size_t)j * H + h * HD + d];
    ctx[(size_t)i * H + h * HD + d] = s * inv;
  }
}

__global__ void add_ln_kernel(const float* __restrict__ x, const float* __restrict__ y,
                              const float* __restrict__ g, const float* __restrict__ b,
                              float* __restrict__ out) {
  const int row = blockIdx.x, tid = threadIdx.x;   // block = 256
  __shared__ float srow[H];
  __shared__ float red[256];
  float psum = 0.f;
  for (int idx = tid; idx < H; idx += 256) {
    float s = x[(size_t)row * H + idx] + y[(size_t)row * H + idx];
    srow[idx] = s; psum += s;
  }
  red[tid] = psum; __syncthreads();
  for (int s = 128; s > 0; s >>= 1) { if (tid < s) red[tid] += red[tid + s]; __syncthreads(); }
  const float mu = red[0] * (1.f / H);
  __syncthreads();
  float pv = 0.f;
  for (int idx = tid; idx < H; idx += 256) { float d = srow[idx] - mu; pv += d * d; }
  red[tid] = pv; __syncthreads();
  for (int s = 128; s > 0; s >>= 1) { if (tid < s) red[tid] += red[tid + s]; __syncthreads(); }
  const float inv = rsqrtf(red[0] * (1.f / H) + 1e-12f);
  for (int idx = tid; idx < H; idx += 256)
    out[(size_t)row * H + idx] = (srow[idx] - mu) * inv * g[idx] + b[idx];
}

__global__ void ms_kernel(const float* __restrict__ x,
                          const float* __restrict__ mW1, const float* __restrict__ mb1,
                          const float* __restrict__ mW2, const float* __restrict__ mb2,
                          const float* __restrict__ mW3, const float* __restrict__ mb3,
                          float* __restrict__ ms) {
  const int row = blockIdx.x, tid = threadIdx.x;   // block = 256
  __shared__ float xr[H];
  __shared__ float h1[HHALF];
  __shared__ float h2[HQ];
  __shared__ float red[256];
  for (int idx = tid; idx < H; idx += 256) xr[idx] = x[(size_t)row * H + idx];
  __syncthreads();
  for (int o = tid; o < HHALF; o += 256) {
    float s = mb1[o];
    for (int hh = 0; hh < H; ++hh) s += xr[hh] * mW1[(size_t)hh * HHALF + o];
    h1[o] = fmaxf(s, 0.f);
  }
  __syncthreads();
  for (int o = tid; o < HQ; o += 256) {
    float s = mb2[o];
    for (int j = 0; j < HHALF; ++j) s += h1[j] * mW2[(size_t)j * HQ + o];
    h2[o] = fmaxf(s, 0.f);
  }
  __syncthreads();
  float p = (tid < HQ) ? h2[tid] * mW3[tid] : 0.f;
  red[tid] = p; __syncthreads();
  for (int s = 128; s > 0; s >>= 1) { if (tid < s) red[tid] += red[tid + s]; __syncthreads(); }
  if (tid == 0) ms[row] = red[0] + mb3[0];
}

__global__ void build_x_kernel(const float* __restrict__ dummy, const float* __restrict__ m,
                               float* __restrict__ x) {
  const int row = blockIdx.x, tid = threadIdx.x;
  const float* src = (row == 0) ? dummy : (m + (size_t)(row - 1) * H);
  for (int idx = tid; idx < H; idx += 256) x[(size_t)row * H + idx] = src[idx];
}

__global__ void cvt_kernel(const float* __restrict__ src, __bf16* __restrict__ dst, int n) {
  const int idx = blockIdx.x * 256 + threadIdx.x;
  if (idx < n) dst[idx] = f2bf(src[idx]);
}

// W [K,N] fp32 -> Wt [N,K] bf16 (so B-fragments load contiguously)
__global__ void tcvt_kernel(const float* __restrict__ W, __bf16* __restrict__ Wt, int K, int N) {
  const int idx = blockIdx.x * 256 + threadIdx.x;
  if (idx >= K * N) return;
  const int k = idx / N, n = idx - k * N;
  Wt[(size_t)n * K + k] = f2bf(W[idx]);
}

// ---------------------------------------------------------------------------
extern "C" void kernel_launch(void* const* d_in, const int* in_sizes, int n_in,
                              void* d_out, int out_size, void* d_ws, size_t ws_size,
                              hipStream_t stream) {
  (void)in_sizes; (void)n_in; (void)out_size; (void)ws_size;

  // ---- inputs (setup_inputs dict order) ----
  const float* seq     = (const float*)d_in[0];
  const float* spk_emb = (const float*)d_in[1];
  const float* dummy   = (const float*)d_in[2];
  const float* Wq  = (const float*)d_in[3];  const float* bq  = (const float*)d_in[4];
  const float* Wk  = (const float*)d_in[5];  const float* bk  = (const float*)d_in[6];
  const float* Wv  = (const float*)d_in[7];  const float* bv  = (const float*)d_in[8];
  const float* Wao = (const float*)d_in[9];  const float* bao = (const float*)d_in[10];
  const float* ln1g = (const float*)d_in[11]; const float* ln1b = (const float*)d_in[12];
  const float* Wi  = (const float*)d_in[13]; const float* bi  = (const float*)d_in[14];
  const float* Wo  = (const float*)d_in[15]; const float* bo  = (const float*)d_in[16];
  const float* ln2g = (const float*)d_in[17]; const float* ln2b = (const float*)d_in[18];
  const float* pW1a = (const float*)d_in[19]; const float* pW1b = (const float*)d_in[20];
  const float* pb1  = (const float*)d_in[21];
  const float* pW2  = (const float*)d_in[22]; const float* pb2 = (const float*)d_in[23];
  const float* pW3  = (const float*)d_in[24]; const float* pb3 = (const float*)d_in[25];
  const float* mW1  = (const float*)d_in[26]; const float* mb1 = (const float*)d_in[27];
  const float* mW2  = (const float*)d_in[28]; const float* mb2 = (const float*)d_in[29];
  const float* mW3  = (const float*)d_in[30]; const float* mb3 = (const float*)d_in[31];
  const int* mseg   = (const int*)d_in[32];
  const int* mstart = (const int*)d_in[33];
  const int* mend   = (const int*)d_in[34];
  const int* spk_ids= (const int*)d_in[35];
  float* out = (float*)d_out;

  // ---- workspace layout ----
  char* w = (char*)d_ws;
  size_t off = 0;
  auto alloc = [&](size_t bytes) -> void* {
    void* p = w + off; off = (off + bytes + 255) & ~(size_t)255; return p;
  };
  const size_t FH  = (size_t)NROWS * H * sizeof(float);      // 512x768 f32
  const size_t BH  = (size_t)NROWS * H * sizeof(__bf16);     // 512x768 bf16
  float*  m_buf = (float*)alloc(FH);
  float*  q_buf = (float*)alloc(FH);
  float*  k_buf = (float*)alloc(FH);
  float*  v_buf = (float*)alloc(FH);
  float*  ctx   = (float*)alloc(FH);
  float*  t0    = (float*)alloc(FH);                          // ao / fo temp
  float*  a_buf = (float*)alloc(FH);
  float*  ff32  = (float*)alloc((size_t)NROWS * FF * sizeof(float));
  float*  pa    = (float*)alloc(FH);
  float*  pb    = (float*)alloc(FH);
  float*  x32   = (float*)alloc(FH);
  float*  msv   = (float*)alloc((size_t)NROWS * sizeof(float));
  __bf16* ab    = (__bf16*)alloc(BH);                         // shared bf16 activ buf
  __bf16* ffb   = (__bf16*)alloc((size_t)NROWS * FF * sizeof(__bf16));
  __bf16* xb    = (__bf16*)alloc(BH);
  __bf16* WqT   = (__bf16*)alloc((size_t)H * H * 2);
  __bf16* WkT   = (__bf16*)alloc((size_t)H * H * 2);
  __bf16* WvT   = (__bf16*)alloc((size_t)H * H * 2);
  __bf16* WaoT  = (__bf16*)alloc((size_t)H * H * 2);
  __bf16* WiT   = (__bf16*)alloc((size_t)H * FF * 2);
  __bf16* WoT   = (__bf16*)alloc((size_t)FF * H * 2);
  __bf16* pW1aT = (__bf16*)alloc((size_t)H * H * 2);
  __bf16* pW1bT = (__bf16*)alloc((size_t)H * H * 2);
  __bf16* pW2T  = (__bf16*)alloc((size_t)H * HHALF * 2);

  auto tcvt = [&](const float* W, __bf16* Wt, int K, int N) {
    int n = K * N;
    tcvt_kernel<<<dim3((n + 255) / 256), dim3(256), 0, stream>>>(W, Wt, K, N);
  };
  auto cvt = [&](const float* src, __bf16* dst, int n) {
    cvt_kernel<<<dim3((n + 255) / 256), dim3(256), 0, stream>>>(src, dst, n);
  };
  auto gemm = [&](const __bf16* A, const __bf16* Bt, const float* bias, float* C,
                  int M, int N, int K, int act) {
    gemm_wmma<<<dim3(N / 64, M / 16), dim3(32), 0, stream>>>(A, Bt, bias, C, M, N, K, act);
  };

  // ---- weight transpose/convert (bf16, [N,K]) ----
  tcvt(Wq,   WqT,   H, H);
  tcvt(Wk,   WkT,   H, H);
  tcvt(Wv,   WvT,   H, H);
  tcvt(Wao,  WaoT,  H, H);
  tcvt(Wi,   WiT,   H, FF);
  tcvt(Wo,   WoT,   FF, H);
  tcvt(pW1a, pW1aT, H, H);
  tcvt(pW1b, pW1bT, H, H);
  tcvt(pW2,  pW2T,  H, HHALF);

  // ---- gather mention reps (pad row 511 zeroed) ----
  gather_kernel<<<dim3(NROWS), dim3(256), 0, stream>>>(seq, spk_emb, mseg, mstart, mend,
                                                       spk_ids, m_buf);

  // ---- two shared BertLayer applications ----
  for (int layer = 0; layer < 2; ++layer) {
    cvt(m_buf, ab, NROWS * H);
    gemm(ab, WqT, bq, q_buf, NROWS, H, H, 0);
    gemm(ab, WkT, bk, k_buf, NROWS, H, H, 0);
    gemm(ab, WvT, bv, v_buf, NROWS, H, H, 0);
    attn_kernel<<<dim3(NH, NROWS), dim3(128), 0, stream>>>(q_buf, k_buf, v_buf, ctx);
    cvt(ctx, ab, NROWS * H);
    gemm(ab, WaoT, bao, t0, NROWS, H, H, 0);
    add_ln_kernel<<<dim3(NROWS), dim3(256), 0, stream>>>(m_buf, t0, ln1g, ln1b, a_buf);
    cvt(a_buf, ab, NROWS * H);
    gemm(ab, WiT, bi, ff32, NROWS, FF, H, /*GELU*/1);
    cvt(ff32, ffb, NROWS * FF);
    gemm(ffb, WoT, bo, t0, NROWS, H, FF, 0);
    add_ln_kernel<<<dim3(NROWS), dim3(256), 0, stream>>>(a_buf, t0, ln2g, ln2b, m_buf);
  }

  // ---- x = concat(dummy, m) ; pa / pb projections ; mention scores ----
  build_x_kernel<<<dim3(NROWS), dim3(256), 0, stream>>>(dummy, m_buf, x32);
  cvt(x32, xb, NROWS * H);
  gemm(xb, pW1aT, nullptr, pa, NROWS, H, H, 0);
  gemm(xb, pW1bT, pb1,     pb, NROWS, H, H, 0);
  ms_kernel<<<dim3(NROWS), dim3(256), 0, stream>>>(x32, mW1, mb1, mW2, mb2, mW3, mb3, msv);

  // ---- fused pairwise scorer -> d_out [512, 512] ----
  pair_kernel<<<dim3(NROWS / 16, NROWS), dim3(32), 0, stream>>>(
      pa, pb, pW2T, pb2, pW3, pb3, msv, out);
}
